// E3DLSTM_57432302682726
// MI455X (gfx1250) — matrix operand order repaired and checked by reading, not compiled, not run on hardware
//
#include <hip/hip_runtime.h>
#include <hip/hip_bf16.h>
#include <stdint.h>

// ---------------------------------------------------------------------------
// E3D-LSTM forward for gfx1250 (MI455X). All GEMM-shaped work (5x5 convs as
// implicit GEMM over taps, flash attention, 1x1 conv) goes through
// v_wmma_f32_16x16x32_bf16 with f32 accumulation.
//
// Conv inputs are maintained as bf16 in a HALOED channel-last layout:
//   [slice][36 rows][36 cols][cpitch]  (2-pixel zero border, cpitch % 32 == 0)
// so the conv inner loop is 4x global_load_b128 + 1 wmma: no converts, no
// bounds cndmasks. Conv weights are pre-packed per-launch into the WMMA
// B-fragment lane layout (two contiguous 16B loads per lane).
// ---------------------------------------------------------------------------

#define HID    64
#define BATCH  2
#define TSEQ   8
#define SLICES 4            // B * WIN
#define POS    1024         // 32*32
#define HROW   36           // 32 + 2*2 halo
#define HSLICE (HROW*HROW)  // 1296
#define QROWS  2048         // HID*WIN*32*32 / 64
#define HIST_MAX 8
#define NELEM64  (SLICES*POS*64)   // 262144 elems of a 64-ch activation

typedef __attribute__((ext_vector_type(16))) __bf16 v16bf;
typedef __attribute__((ext_vector_type(8)))  float  v8f;

struct H16 { unsigned short h[16]; };            // 32 bytes == one v16bf

__device__ static inline v16bf h2v(H16 x) { return __builtin_bit_cast(v16bf, x); }

__device__ static inline unsigned short f2bf(float f) {
  unsigned int u = __float_as_uint(f);
  u += 0x7FFFu + ((u >> 16) & 1u);               // round-to-nearest-even
  return (unsigned short)(u >> 16);
}
__device__ static inline float sigm(float x) { return 1.0f / (1.0f + __expf(-x)); }

// haloed bf16 index for interior pixel (y,x in 0..31)
__device__ static inline size_t hidx(int slice, int y, int x, int pitch, int c) {
  return ((size_t)slice * HSLICE + (size_t)(y + 2) * HROW + (x + 2)) * pitch + c;
}

// ---------------------------------------------------------------------------
// Weight packing: w (Cout, Cin, 1, k, k) f32  ->  bf16 B-fragment tiles
// layout: [tap][ktile][ntile][lane(32)][half(16)]
//   lane<16 : N = ntile*16+lane,    K = ktile*32 + 0..15  (halves 0..15)
//   lane>=16: N = ntile*16+lane-16, K = ktile*32 + 16..31
// ---------------------------------------------------------------------------
__global__ void pack_w(const float* __restrict__ w, unsigned short* __restrict__ out,
                       int cin, int cout, int ksz) {
  int ktiles = (cin + 31) >> 5, ntiles = cout >> 4, taps = ksz * ksz;
  size_t total = (size_t)taps * ktiles * ntiles * 512;
  size_t idx = (size_t)blockIdx.x * blockDim.x + threadIdx.x;
  if (idx >= total) return;
  int h    = (int)(idx & 15);
  int lane = (int)((idx >> 4) & 31);
  size_t tile = idx >> 9;
  int nt  = (int)(tile % ntiles);
  int kt  = (int)((tile / ntiles) % ktiles);
  int tap = (int)(tile / ((size_t)ntiles * ktiles));
  int n = nt * 16 + (lane & 15);
  int k = kt * 32 + (lane >> 4) * 16 + h;
  float v = 0.0f;
  if (k < cin) {
    int ky = tap / ksz, kx = tap % ksz;
    v = w[(((size_t)n * cin + k) * ksz + ky) * ksz + kx];
  }
  out[idx] = f2bf(v);
}

// ---------------------------------------------------------------------------
// Implicit-GEMM conv over a haloed bf16 input. One wave computes a
// 16(M=spatial)x16(N=cout) f32 tile; loops taps x ktiles WMMA k-steps.
// ---------------------------------------------------------------------------
__global__ void conv_gemm(const unsigned short* __restrict__ in, int cpitch, int ktiles,
                          const unsigned short* __restrict__ wpak,
                          const float* __restrict__ bias,
                          float* __restrict__ out, int cout, int ksz) {
  int ntiles = cout >> 4;
  int mt = blockIdx.x % 64;        // 64 m-tiles per 32x32 slice
  int nt = blockIdx.x / 64;
  int slice = blockIdx.y;
  int lane = threadIdx.x;
  int y  = mt >> 1;
  int xb = (mt & 1) << 4;
  int mrow = lane & 15;
  int hi   = lane >> 4;
  int x = xb + mrow;               // A-fragment row position (M = mrow)
  int pad = ksz >> 1;
  const unsigned short* inS = in + (size_t)slice * HSLICE * cpitch;
  v8f acc = {};
  for (int ky = 0; ky < ksz; ++ky) {
    int row = y + ky - pad + 2;    // haloed row, always in [0,35]
    for (int kx = 0; kx < ksz; ++kx) {
      int col = x + kx - pad + 2;
      const unsigned short* p = inS + (size_t)(row * HROW + col) * cpitch;
      int tap = ky * ksz + kx;
      for (int kt = 0; kt < ktiles; ++kt) {
        int kA = kt * 32 + hi * 16;
        H16 a;
        *(uint4*)&a.h[0] = *(const uint4*)&p[kA];        // K 0..7 / 16..23
        *(uint4*)&a.h[8] = *(const uint4*)&p[kA + 8];    // K 8..15 / 24..31
        const unsigned short* wt =
            wpak + ((((size_t)tap * ktiles + kt) * ntiles + nt) * 512) + lane * 16;
        __builtin_prefetch(wt + 512, 0, 3);              // global_prefetch next tile
        H16 b;
        *(uint4*)&b.h[0] = *(const uint4*)&wt[0];
        *(uint4*)&b.h[8] = *(const uint4*)&wt[8];
        acc = __builtin_amdgcn_wmma_f32_16x16x32_bf16(
            false, h2v(a), false, h2v(b), (short)0, acc, false, false);
      }
    }
  }
  float bv = bias[nt * 16 + mrow];
  float* o = out + ((size_t)slice * POS + (size_t)y * 32) * cout + nt * 16 + mrow;
#pragma unroll
  for (int v = 0; v < 8; ++v) {
    int M = v + 8 * hi;
    o[(size_t)(xb + M) * cout] = acc[v] + bv;
  }
}

// ---------------------------------------------------------------------------
// Flash attention, one wave per 16 q-rows. qf: bf16 (B,2048,64) row-major.
// histL: bf16 (B, HIST_MAX*2048, 64). Streams 32 kf rows/iter through LDS.
// ---------------------------------------------------------------------------
__global__ void attn_flash(const unsigned short* __restrict__ qf,
                           const unsigned short* __restrict__ histL,
                           float* __restrict__ attn, int Yrows) {
  __shared__ __align__(16) unsigned short kbuf[32 * 64];   // row-major rows
  __shared__ __align__(16) unsigned short ktbuf[64 * 32];  // transposed (ch, row)
  __shared__ __align__(16) unsigned short pbuf[16 * 32];   // P tile bf16
  int b = blockIdx.y, qt = blockIdx.x, lane = threadIdx.x;
  int mrow = lane & 15, hi = lane >> 4;
  const unsigned short* qB = qf + (size_t)b * QROWS * 64;
  const unsigned short* kB = histL + (size_t)b * HIST_MAX * QROWS * 64;
  int qrow = qt * 16 + mrow;
  H16 aq0, aq1;
  {
    const unsigned short* q0 = qB + (size_t)qrow * 64;
    *(uint4*)&aq0.h[0] = *(const uint4*)&q0[hi * 16];
    *(uint4*)&aq0.h[8] = *(const uint4*)&q0[hi * 16 + 8];
    *(uint4*)&aq1.h[0] = *(const uint4*)&q0[32 + hi * 16];
    *(uint4*)&aq1.h[8] = *(const uint4*)&q0[32 + hi * 16 + 8];
  }
  v8f occ[4]; occ[0] = {}; occ[1] = {}; occ[2] = {}; occ[3] = {};
  float mrw[8], lrw[8];
#pragma unroll
  for (int v = 0; v < 8; ++v) { mrw[v] = -3.0e38f; lrw[v] = 0.0f; }

  for (int yb = 0; yb < Yrows; yb += 32) {
    // cooperative load of 32 kf rows: row-major + transposed staging
    const unsigned short* krow = kB + (size_t)(yb + lane) * 64;
#pragma unroll
    for (int j = 0; j < 8; ++j) {
      uint4 u = ((const uint4*)krow)[j];
      *(uint4*)&kbuf[lane * 64 + j * 8] = u;
      unsigned int wds[4] = {u.x, u.y, u.z, u.w};
#pragma unroll
      for (int q = 0; q < 4; ++q) {
        int c = j * 8 + q * 2;
        ktbuf[(c)     * 32 + lane] = (unsigned short)(wds[q] & 0xFFFFu);
        ktbuf[(c + 1) * 32 + lane] = (unsigned short)(wds[q] >> 16);
      }
    }
    __syncthreads();

    // S tiles: S = q(16x64) . kf^T(64x32), two 16-col halves
    v8f sT[2];
#pragma unroll
    for (int nh = 0; nh < 2; ++nh) {
      int lrowi = nh * 16 + mrow;                 // kf local row (S column)
      H16 b0, b1;
      *(uint4*)&b0.h[0] = *(const uint4*)&kbuf[lrowi * 64 + hi * 16];
      *(uint4*)&b0.h[8] = *(const uint4*)&kbuf[lrowi * 64 + hi * 16 + 8];
      *(uint4*)&b1.h[0] = *(const uint4*)&kbuf[lrowi * 64 + 32 + hi * 16];
      *(uint4*)&b1.h[8] = *(const uint4*)&kbuf[lrowi * 64 + 32 + hi * 16 + 8];
      v8f s = {};
      s = __builtin_amdgcn_wmma_f32_16x16x32_bf16(false, h2v(aq0), false, h2v(b0),
                                                  (short)0, s, false, false);
      s = __builtin_amdgcn_wmma_f32_16x16x32_bf16(false, h2v(aq1), false, h2v(b1),
                                                  (short)0, s, false, false);
      sT[nh] = s;
    }

    // online softmax: row stats via xor-shuffles within each 16-lane half
#pragma unroll
    for (int v = 0; v < 8; ++v) {
      float mx = fmaxf(sT[0][v], sT[1][v]);
      mx = fmaxf(mx, __shfl_xor(mx, 1, 32));
      mx = fmaxf(mx, __shfl_xor(mx, 2, 32));
      mx = fmaxf(mx, __shfl_xor(mx, 4, 32));
      mx = fmaxf(mx, __shfl_xor(mx, 8, 32));
      float mn = fmaxf(mrw[v], mx);
      float sc = __expf(mrw[v] - mn);
      float p0 = __expf(sT[0][v] - mn);
      float p1 = __expf(sT[1][v] - mn);
      float rs = p0 + p1;
      rs += __shfl_xor(rs, 1, 32);
      rs += __shfl_xor(rs, 2, 32);
      rs += __shfl_xor(rs, 4, 32);
      rs += __shfl_xor(rs, 8, 32);
      lrw[v] = lrw[v] * sc + rs;
      mrw[v] = mn;
      occ[0][v] *= sc; occ[1][v] *= sc; occ[2][v] *= sc; occ[3][v] *= sc;
      sT[0][v] = p0; sT[1][v] = p1;
    }

    // stage P (16x32) as bf16, reload in A-fragment layout
#pragma unroll
    for (int v = 0; v < 8; ++v) {
      pbuf[(v + 8 * hi) * 32 + mrow]      = f2bf(sT[0][v]);
      pbuf[(v + 8 * hi) * 32 + 16 + mrow] = f2bf(sT[1][v]);
    }
    __syncthreads();
    H16 ap;
    *(uint4*)&ap.h[0] = *(const uint4*)&pbuf[mrow * 32 + hi * 16];
    *(uint4*)&ap.h[8] = *(const uint4*)&pbuf[mrow * 32 + hi * 16 + 8];

    // O(16x64) += P(16x32) . kf(32x64), B fragments from transposed staging
#pragma unroll
    for (int nt2 = 0; nt2 < 4; ++nt2) {
      int ch = nt2 * 16 + mrow;
      H16 bf2;
      *(uint4*)&bf2.h[0] = *(const uint4*)&ktbuf[ch * 32 + hi * 16];
      *(uint4*)&bf2.h[8] = *(const uint4*)&ktbuf[ch * 32 + hi * 16 + 8];
      occ[nt2] = __builtin_amdgcn_wmma_f32_16x16x32_bf16(
          false, h2v(ap), false, h2v(bf2), (short)0, occ[nt2], false, false);
    }
    __syncthreads();
  }

  float* op = attn + ((size_t)b * QROWS + (size_t)qt * 16) * 64;
#pragma unroll
  for (int nt2 = 0; nt2 < 4; ++nt2) {
#pragma unroll
    for (int v = 0; v < 8; ++v) {
      int M = v + 8 * hi;
      op[(size_t)M * 64 + nt2 * 16 + mrow] = occ[nt2][v] / lrw[v];
    }
  }
}

// ---------------------------------------------------------------------------
// Elementwise / glue kernels. Channel-last index: i -> pos = i>>6, c = i&63,
// pos = slice*1024 + y*32 + x, slice = b*2 + d.
// NCDHW-reshape remap used by attention buffers:
//   xi = c*32 + d*16 + (y>>1),  c2 = ((y&1)<<5) | x
// ---------------------------------------------------------------------------
__global__ void build_xwin(const float* __restrict__ input_x,
                           unsigned short* __restrict__ xwin_bf, int t) {
  int i = blockIdx.x * blockDim.x + threadIdx.x;
  if (i >= SLICES * POS * 16) return;
  int c = i & 15; int pos = i >> 4;
  int slice = pos >> 10, b = slice >> 1, d = slice & 1;
  int yx = pos & 1023, y = yx >> 5, x = yx & 31;
  int tt = t - 1 + d;
  float v = (tt < 0) ? 0.0f : input_x[(((size_t)b * TSEQ + tt) * 16 + c) * POS + yx];
  xwin_bf[hidx(slice, y, x, 32, c)] = f2bf(v);   // chans 16..31 stay zero
}

__global__ void hist_append(const float* __restrict__ cst,
                            unsigned short* __restrict__ histL, int entry) {
  int i = blockIdx.x * blockDim.x + threadIdx.x;
  if (i >= NELEM64) return;
  int c = i & 63; int pos = i >> 6;
  int slice = pos >> 10, b = slice >> 1, d = slice & 1;
  int y = (pos >> 5) & 31, x = pos & 31;
  int xi = c * 32 + d * 16 + (y >> 1);
  int c2 = ((y & 1) << 5) | x;
  histL[((size_t)b * HIST_MAX * QROWS + (size_t)entry * QROWS + xi) * 64 + c2] = f2bf(cst[i]);
}

__global__ void gates1(const float* __restrict__ xcat, const float* __restrict__ hcat,
                       float* __restrict__ it, float* __restrict__ gt,
                       unsigned short* __restrict__ qf) {
  int i = blockIdx.x * blockDim.x + threadIdx.x;
  if (i >= NELEM64) return;
  int c = i & 63; int pos = i >> 6;
  const float* xr = xcat + (size_t)pos * 448;
  const float* hr = hcat + (size_t)pos * 256;
  float i_t = sigm(xr[c] + hr[c]);
  float g_t = tanhf(xr[64 + c] + hr[64 + c]);
  float r_t = sigm(xr[128 + c] + hr[128 + c]);
  it[i] = i_t; gt[i] = g_t;
  int slice = pos >> 10, b = slice >> 1, d = slice & 1;
  int y = (pos >> 5) & 31, x = pos & 31;
  int xi = c * 32 + d * 16 + (y >> 1);
  int c2 = ((y & 1) << 5) | x;
  qf[((size_t)b * QROWS + xi) * 64 + c2] = f2bf(r_t);
}

__global__ void cellpre(const float* __restrict__ cst, const float* __restrict__ attn,
                        float* __restrict__ pre) {
  int i = blockIdx.x * blockDim.x + threadIdx.x;
  if (i >= NELEM64) return;
  int c = i & 63; int pos = i >> 6;
  int slice = pos >> 10, b = slice >> 1, d = slice & 1;
  int y = (pos >> 5) & 31, x = pos & 31;
  int xi = c * 32 + d * 16 + (y >> 1);
  int c2 = ((y & 1) << 5) | x;
  pre[i] = cst[i] + attn[((size_t)b * QROWS + xi) * 64 + c2];
}

__global__ void bstats(const float* __restrict__ pre, float* __restrict__ stats) {
  __shared__ float s1[256], s2[256];
  int b = blockIdx.x;
  const float* p = pre + (size_t)b * 131072;
  float a = 0.0f, q = 0.0f;
  for (int i = threadIdx.x; i < 131072; i += 256) { float v = p[i]; a += v; q += v * v; }
  s1[threadIdx.x] = a; s2[threadIdx.x] = q; __syncthreads();
  for (int s = 128; s > 0; s >>= 1) {
    if ((int)threadIdx.x < s) { s1[threadIdx.x] += s1[threadIdx.x + s]; s2[threadIdx.x] += s2[threadIdx.x + s]; }
    __syncthreads();
  }
  if (threadIdx.x == 0) {
    float m = s1[0] * (1.0f / 131072.0f);
    float var = s2[0] * (1.0f / 131072.0f) - m * m;
    stats[b * 2] = m; stats[b * 2 + 1] = rsqrtf(var + 1e-5f);
  }
}

__global__ void cellfinal(const float* __restrict__ pre, const float* __restrict__ stats,
                          const float* __restrict__ it, const float* __restrict__ gt,
                          float* __restrict__ cst, unsigned short* __restrict__ ncg_bf) {
  int i = blockIdx.x * blockDim.x + threadIdx.x;
  if (i >= NELEM64) return;
  int c = i & 63; int pos = i >> 6;
  int slice = pos >> 10, b = pos >> 11;
  int y = (pos >> 5) & 31, x = pos & 31;
  float nc = (pre[i] - stats[b * 2]) * stats[b * 2 + 1] + it[i] * gt[i];
  cst[i] = nc;
  ncg_bf[hidx(slice, y, x, 128, c)] = f2bf(nc);
}

__global__ void gmgates(const float* __restrict__ gmc, const float* __restrict__ xcat,
                        float* __restrict__ gmem, unsigned short* __restrict__ gm_bf,
                        unsigned short* __restrict__ ncg_bf) {
  int i = blockIdx.x * blockDim.x + threadIdx.x;
  if (i >= NELEM64) return;
  int c = i & 63; int pos = i >> 6;
  int slice = pos >> 10, y = (pos >> 5) & 31, x = pos & 31;
  const float* g  = gmc  + (size_t)pos * 256;
  const float* xr = xcat + (size_t)pos * 448;
  float ti = sigm(xr[192 + c] + g[c]);             // i_xp + i_m
  float tf = sigm(xr[320 + c] + g[64 + c] + 1.0f); // f_xp + f_m + FORGET_BIAS
  float tg = tanhf(xr[256 + c] + g[128 + c]);      // g_xp + g_m
  float ngm = tf * tanhf(g[192 + c]) + ti * tg;    // tanh(m_m)
  gmem[i] = ngm;
  unsigned short bv = f2bf(ngm);
  gm_bf[hidx(slice, y, x, 64, c)] = bv;
  ncg_bf[hidx(slice, y, x, 128, 64 + c)] = bv;
}

__global__ void houtk(const float* __restrict__ xcat, const float* __restrict__ hcat,
                      const float* __restrict__ oc, const float* __restrict__ om,
                      const float* __restrict__ memo, float* __restrict__ h,
                      unsigned short* __restrict__ h_bf) {
  int i = blockIdx.x * blockDim.x + threadIdx.x;
  if (i >= NELEM64) return;
  int c = i & 63; int pos = i >> 6;
  int slice = pos >> 10, y = (pos >> 5) & 31, x = pos & 31;
  float og = tanhf(xcat[(size_t)pos * 448 + 384 + c] + hcat[(size_t)pos * 256 + 192 + c]
                   + oc[i] + om[i]);
  float hv = tanhf(memo[i]) * sigm(og);
  h[i] = hv;
  h_bf[hidx(slice, y, x, 64, c)] = f2bf(hv);
}

__global__ void out_frame(const float* __restrict__ h1, const float* __restrict__ wout,
                          float* __restrict__ nf, int t) {
  int i = blockIdx.x * blockDim.x + threadIdx.x;
  if (i >= BATCH * 16 * POS) return;
  int yx = i & 1023, o = (i >> 10) & 15, b = i >> 14;
  float acc = 0.0f;
#pragma unroll
  for (int d = 0; d < 2; ++d) {
    const float* hp = h1 + (((size_t)(b * 2 + d) * POS) + yx) * 64;
    for (int c = 0; c < 64; ++c) acc += hp[c] * wout[((size_t)o * 64 + c) * 2 + d];
  }
  nf[(((size_t)b * TSEQ + t) * 16 + o) * POS + yx] = fminf(fmaxf(acc, 0.0f), 1.0f);
}

__global__ void loss_k(const float* __restrict__ nf, const float* __restrict__ tgt,
                       float* __restrict__ lossout) {
  __shared__ float sa[256], sb[256];
  float m = 0.0f, a = 0.0f;
  const int N = BATCH * TSEQ * 16 * POS;
  for (int i = threadIdx.x; i < N; i += 256) {
    float d = nf[i] - tgt[i]; m += d * d; a += fabsf(d);
  }
  sa[threadIdx.x] = m; sb[threadIdx.x] = a; __syncthreads();
  for (int s = 128; s > 0; s >>= 1) {
    if ((int)threadIdx.x < s) { sa[threadIdx.x] += sa[threadIdx.x + s]; sb[threadIdx.x] += sb[threadIdx.x + s]; }
    __syncthreads();
  }
  if (threadIdx.x == 0) lossout[0] = sa[0] / (float)N + sb[0] / (float)N;
}

// ---------------------------------------------------------------------------
// Host orchestration
// ---------------------------------------------------------------------------
extern "C" void kernel_launch(void* const* d_in, const int* in_sizes, int n_in,
                              void* d_out, int out_size, void* d_ws, size_t ws_size,
                              hipStream_t stream) {
  (void)in_sizes; (void)n_in; (void)out_size; (void)ws_size;
  const float* input_x = (const float*)d_in[0];
  const float* targets = (const float*)d_in[1];
  // Params: flattened by sorted dict keys: cell{bc,bg,bh,bl,bm,bx,wc,wg,wh,wl,wm,wx}, then w_out
  struct Cell { const float *bc,*bg,*bh,*bl,*bm,*bx,*wc,*wg,*wh,*wl,*wm,*wx; };
  Cell cell[2];
  for (int i = 0; i < 2; ++i) {
    int b = 2 + i * 12;
    cell[i].bc = (const float*)d_in[b + 0];  cell[i].bg = (const float*)d_in[b + 1];
    cell[i].bh = (const float*)d_in[b + 2];  cell[i].bl = (const float*)d_in[b + 3];
    cell[i].bm = (const float*)d_in[b + 4];  cell[i].bx = (const float*)d_in[b + 5];
    cell[i].wc = (const float*)d_in[b + 6];  cell[i].wg = (const float*)d_in[b + 7];
    cell[i].wh = (const float*)d_in[b + 8];  cell[i].wl = (const float*)d_in[b + 9];
    cell[i].wm = (const float*)d_in[b + 10]; cell[i].wx = (const float*)d_in[b + 11];
  }
  const float* w_out = (const float*)d_in[26];

  // ---- workspace carve ----
  char* ws = (char*)d_ws; size_t off = 0;
  auto allocF = [&](size_t n) { float* p = (float*)(ws + off); off += ((n * 4 + 255) & ~(size_t)255); return p; };
  auto allocH = [&](size_t n) { unsigned short* p = (unsigned short*)(ws + off); off += ((n * 2 + 255) & ~(size_t)255); return p; };

  float* hS[2] = { allocF(NELEM64), allocF(NELEM64) };
  float* cS[2] = { allocF(NELEM64), allocF(NELEM64) };
  float* gmem  = allocF(NELEM64);
  float* xcat  = allocF((size_t)SLICES * POS * 448);
  float* hcat  = allocF((size_t)SLICES * POS * 256);
  float* gmc   = allocF((size_t)SLICES * POS * 256);
  float* itb   = allocF(NELEM64);
  float* gtb   = allocF(NELEM64);
  float* pre   = allocF(NELEM64);
  float* ocb   = allocF(NELEM64);
  float* omb   = allocF(NELEM64);
  float* memo  = allocF(NELEM64);
  float* attn  = allocF((size_t)BATCH * QROWS * 64);
  float* stats = allocF(8);
  // haloed bf16 conv inputs
  unsigned short* xwin_bf = allocH((size_t)SLICES * HSLICE * 32);
  unsigned short* h_bf[2] = { allocH((size_t)SLICES * HSLICE * 64),
                              allocH((size_t)SLICES * HSLICE * 64) };
  unsigned short* gm_bf   = allocH((size_t)SLICES * HSLICE * 64);
  unsigned short* ncg_bf  = allocH((size_t)SLICES * HSLICE * 128);
  unsigned short* qf = allocH((size_t)BATCH * QROWS * 64);
  unsigned short* hist[2] = { allocH((size_t)BATCH * HIST_MAX * QROWS * 64),
                              allocH((size_t)BATCH * HIST_MAX * QROWS * 64) };
  auto pakHalfs = [](int cin, int cout, int k) {
    return (size_t)(k * k) * ((cin + 31) / 32) * (cout / 16) * 512;
  };
  unsigned short* wx_p[2] = { allocH(pakHalfs(16, 448, 5)), allocH(pakHalfs(64, 448, 5)) };
  unsigned short* wh_p[2] = { allocH(pakHalfs(64, 256, 5)), allocH(pakHalfs(64, 256, 5)) };
  unsigned short* wm_p[2] = { allocH(pakHalfs(64, 256, 5)), allocH(pakHalfs(64, 256, 5)) };
  unsigned short* wc_p[2] = { allocH(pakHalfs(64, 64, 5)),  allocH(pakHalfs(64, 64, 5)) };
  unsigned short* wg_p[2] = { allocH(pakHalfs(64, 64, 5)),  allocH(pakHalfs(64, 64, 5)) };
  unsigned short* wl_p[2] = { allocH(pakHalfs(128, 64, 1)), allocH(pakHalfs(128, 64, 1)) };

  // ---- zero recurrent state + bf16 halos (ws is NOT re-poisoned between calls)
  hipMemsetAsync(hS[0], 0, NELEM64 * 4, stream);
  hipMemsetAsync(hS[1], 0, NELEM64 * 4, stream);
  hipMemsetAsync(cS[0], 0, NELEM64 * 4, stream);
  hipMemsetAsync(cS[1], 0, NELEM64 * 4, stream);
  hipMemsetAsync(gmem,  0, NELEM64 * 4, stream);
  hipMemsetAsync(xwin_bf, 0, (size_t)SLICES * HSLICE * 32 * 2, stream);
  hipMemsetAsync(h_bf[0], 0, (size_t)SLICES * HSLICE * 64 * 2, stream);
  hipMemsetAsync(h_bf[1], 0, (size_t)SLICES * HSLICE * 64 * 2, stream);
  hipMemsetAsync(gm_bf,   0, (size_t)SLICES * HSLICE * 64 * 2, stream);
  hipMemsetAsync(ncg_bf,  0, (size_t)SLICES * HSLICE * 128 * 2, stream);

  auto pack = [&](const float* w, unsigned short* dst, int cin, int cout, int k) {
    size_t total = pakHalfs(cin, cout, k);
    pack_w<<<dim3((unsigned)((total + 255) / 256)), dim3(256), 0, stream>>>(w, dst, cin, cout, k);
  };
  for (int i = 0; i < 2; ++i) {
    pack(cell[i].wx, wx_p[i], i == 0 ? 16 : 64, 448, 5);
    pack(cell[i].wh, wh_p[i], 64, 256, 5);
    pack(cell[i].wm, wm_p[i], 64, 256, 5);
    pack(cell[i].wc, wc_p[i], 64, 64, 5);
    pack(cell[i].wg, wg_p[i], 64, 64, 5);
    pack(cell[i].wl, wl_p[i], 128, 64, 1);
  }

  auto conv = [&](const unsigned short* in_bf, int cpitch, int ktiles,
                  const unsigned short* wp, const float* bias,
                  float* out, int cout, int k) {
    conv_gemm<<<dim3(64 * (cout / 16), SLICES), dim3(32), 0, stream>>>(
        in_bf, cpitch, ktiles, wp, bias, out, cout, k);
  };
  auto ew = [&](int n) { return dim3((unsigned)((n + 255) / 256)); };

  float* nf = (float*)d_out;
  for (int t = 0; t < TSEQ; ++t) {
    build_xwin<<<ew(SLICES * POS * 16), 256, 0, stream>>>(input_x, xwin_bf, t);
    for (int i = 0; i < 2; ++i) {
      // append pre-update cell state to history (entry t; zeros at t=0)
      hist_append<<<ew(NELEM64), 256, 0, stream>>>(cS[i], hist[i], t);
      // xcat / hcat convolutions
      if (i == 0) conv(xwin_bf, 32, 1, wx_p[0], cell[0].bx, xcat, 448, 5);
      else        conv(h_bf[0], 64, 2, wx_p[1], cell[1].bx, xcat, 448, 5);
      conv(h_bf[i], 64, 2, wh_p[i], cell[i].bh, hcat, 256, 5);
      // gates + q repack
      gates1<<<ew(NELEM64), 256, 0, stream>>>(xcat, hcat, itb, gtb, qf);
      // attention over history (Y = (t+1)*2048 rows)
      attn_flash<<<dim3(QROWS / 16, BATCH), dim3(32), 0, stream>>>(
          qf, hist[i], attn, (t + 1) * QROWS);
      // new_cell = layernorm(c + attn) + i*g
      cellpre<<<ew(NELEM64), 256, 0, stream>>>(cS[i], attn, pre);
      bstats<<<dim3(BATCH), dim3(256), 0, stream>>>(pre, stats);
      cellfinal<<<ew(NELEM64), 256, 0, stream>>>(pre, stats, itb, gtb, cS[i], ncg_bf);
      // global memory branch
      conv(gm_bf, 64, 2, wm_p[i], cell[i].bm, gmc, 256, 5);
      gmgates<<<ew(NELEM64), 256, 0, stream>>>(gmc, xcat, gmem, gm_bf, ncg_bf);
      // output gate convs + 1x1 mem conv (ncg_bf holds [new_cell | new_gm], pitch 128)
      conv(ncg_bf,      128, 2, wc_p[i], cell[i].bc, ocb,  64, 5);
      conv(ncg_bf + 64, 128, 2, wg_p[i], cell[i].bg, omb,  64, 5);
      conv(ncg_bf,      128, 4, wl_p[i], cell[i].bl, memo, 64, 1);
      houtk<<<ew(NELEM64), 256, 0, stream>>>(xcat, hcat, ocb, omb, memo, hS[i], h_bf[i]);
    }
    out_frame<<<ew(BATCH * 16 * POS), 256, 0, stream>>>(hS[1], w_out, nf, t);
  }
  loss_k<<<dim3(1), dim3(256), 0, stream>>>(nf, targets, nf + BATCH * TSEQ * 16 * POS);
}
